// UAFM_77567109366450
// MI455X (gfx1250) — matrix operand order repaired and checked by reading, not compiled, not run on hardware
//
#include <hip/hip_runtime.h>
#include <hip/hip_bf16.h>

typedef __attribute__((ext_vector_type(16))) _Float16 v16h;
typedef __attribute__((ext_vector_type(8)))  _Float16 v8h;
typedef __attribute__((ext_vector_type(8)))  float    v8f;

#define HH 128
#define WW 128
#define HW (HH*WW)   // 16384
#define NB 2
#define NTILES (NB*HH*(WW/16))   // 2048

// ---------------- WMMA helpers (16x16x32 f16 -> f32) ----------------
// A-matrix 16x32 f16 layout: lane l: row M = l&15; hi = l>>4.
//   elems 0..15 -> K = k0 + hi*16 + i
__device__ __forceinline__ v16h load_a_frag(const _Float16* Arow, int k0, int hi) {
    v8h lo = *(const v8h*)(Arow + k0 + hi * 16);
    v8h hi8 = *(const v8h*)(Arow + k0 + hi * 16 + 8);
    v16h a;
#pragma unroll
    for (int i = 0; i < 8; ++i) { a[i] = lo[i]; a[i + 8] = hi8[i]; }
    return a;
}
// B-matrix 32x16 f16 layout: lane l: col N = l&15; elems i -> K = k0 + (l>>4)*16 + i
// (contiguous -> direct v16h load from [n][k]-ordered LDS)

__device__ __forceinline__ v8f wmma_step(v16h a, v16h b, v8f c) {
    return __builtin_amdgcn_wmma_f32_16x16x32_f16(false, a, false, b, (short)0, c,
                                                  false, false);
}

// ---------------- prep: f32 -> f16 weight conversion ----------------
__global__ void k_f32_to_f16(const float* __restrict__ src, _Float16* __restrict__ dst, int n) {
    int i = blockIdx.x * blockDim.x + threadIdx.x;
    if (i < n) dst[i] = (_Float16)src[i];
}

// wm[t][o][c] = w_out[o][c] * masks[t][c]
__global__ void k_wm(const float* __restrict__ w_out, const int* __restrict__ masks,
                     float* __restrict__ wm) {
    int i = blockIdx.x * blockDim.x + threadIdx.x;
    if (i < 10 * 4 * 128) {
        int c = i & 127;
        int o = (i >> 7) & 3;
        int t = i >> 9;
        wm[i] = w_out[o * 128 + c] * (float)masks[t * 128 + c];
    }
}

// ---------------- x = conv3x3(F_1, w_uec) + b : implicit GEMM, WMMA ----------------
// M=128 (8 waves * 16), K=576, N=16 pixels per block
__global__ __launch_bounds__(256) void k_conv_uec(const float* __restrict__ F1,
                                                  const _Float16* __restrict__ wA,
                                                  const float* __restrict__ bias,
                                                  float* __restrict__ xout) {
    int tile = blockIdx.x;
    int b = tile >> 10;
    int r = tile & 1023;
    int h = r >> 3;
    int w0 = (r & 7) << 4;

    __shared__ __attribute__((aligned(32))) _Float16 ldsB[16][576];

    const float* Fb = F1 + (size_t)b * 64 * HW;
    for (int idx = threadIdx.x; idx < 16 * 576; idx += 256) {
        int n = idx / 576;
        int k = idx - n * 576;
        int c = k / 9;
        int j = k - c * 9;
        int dy = j / 3 - 1, dx = j % 3 - 1;
        int hh = h + dy, wx = w0 + n + dx;
        float v = 0.f;
        if (hh >= 0 && hh < HH && wx >= 0 && wx < WW) v = Fb[c * HW + hh * WW + wx];
        ldsB[n][k] = (_Float16)v;
    }
    __syncthreads();

    int wave = threadIdx.x >> 5;
    int lane = threadIdx.x & 31;
    int lm = lane & 15;
    int hi = lane >> 4;

    v8f acc = {};
    const _Float16* Arow = wA + (size_t)(wave * 16 + lm) * 576;
    for (int k0 = 0; k0 < 576; k0 += 32) {
        v16h a = load_a_frag(Arow, k0, hi);
        v16h bf = *(const v16h*)&ldsB[lm][k0 + hi * 16];
        acc = wmma_step(a, bf, acc);
    }
#pragma unroll
    for (int r2 = 0; r2 < 8; ++r2) {
        int oc = wave * 16 + hi * 8 + r2;
        xout[((size_t)(b * 128 + oc)) * HW + h * WW + w0 + lm] = acc[r2] + bias[oc];
    }
}

// ---------------- ensemble stats: EU (var, ddof=1), mean ----------------
__global__ void k_stats(const float* __restrict__ x, const float* __restrict__ ms,
                        const float* __restrict__ wm, const float* __restrict__ b_out,
                        float* __restrict__ EU_out, float* __restrict__ mean_out,
                        float* __restrict__ EU_ws) {
    int p = blockIdx.x * blockDim.x + threadIdx.x;
    if (p >= NB * HW) return;
    int b = p / HW;
    int hw = p - b * HW;

    float acc[10][4] = {};
    const float* xb = x + (size_t)b * 128 * HW + hw;
    for (int c = 0; c < 128; ++c) {
        float xv = xb[(size_t)c * HW];
#pragma unroll
        for (int t = 0; t < 10; ++t)
#pragma unroll
            for (int o = 0; o < 4; ++o)
                acc[t][o] += wm[(t * 4 + o) * 128 + c] * xv;
    }
#pragma unroll
    for (int o = 0; o < 4; ++o) {
        float msv = ms[(size_t)(b * 4 + o) * HW + hw];
        float bo = b_out[o];
        float s = 0.f, ss = 0.f;
#pragma unroll
        for (int t = 0; t < 10; ++t) {
            float v = tanhf(acc[t][o] + bo) + msv;
            s += v;
            ss += v * v;
        }
        float mean = s * 0.1f;
        float var = (ss - 10.f * mean * mean) * (1.f / 9.f);
        size_t oidx = (size_t)(b * 4 + o) * HW + hw;
        mean_out[oidx] = mean;
        EU_out[oidx] = var;
        EU_ws[oidx] = var;
    }
}

// ---------------- AU = sigmoid(conv3x3(x, w_aue) + b) ----------------
__global__ void k_au(const float* __restrict__ x, const float* __restrict__ w_aue,
                     const float* __restrict__ b_aue, float* __restrict__ AU) {
    int idx = blockIdx.x * blockDim.x + threadIdx.x;
    if (idx >= NB * 4 * HW) return;
    int hw = idx & (HW - 1);
    int bo = idx >> 14;
    int b = bo >> 2, o = bo & 3;
    int h = hw >> 7, w = hw & 127;
    float s = b_aue[o];
    for (int c = 0; c < 128; ++c) {
        const float* xc = x + (size_t)(b * 128 + c) * HW;
        const float* wr = w_aue + (o * 128 + c) * 9;
#pragma unroll
        for (int dy = -1; dy <= 1; ++dy) {
            int hh = h + dy;
            if (hh < 0 || hh >= HH) continue;
#pragma unroll
            for (int dx = -1; dx <= 1; ++dx) {
                int wx = w + dx;
                if (wx < 0 || wx >= WW) continue;
                s += wr[(dy + 1) * 3 + (dx + 1)] * xc[hh * WW + wx];
            }
        }
    }
    AU[idx] = 1.f / (1.f + expf(-s));
}

// ---------------- spectral (sp, WMMA) + spatial (sa) attention ----------------
__global__ __launch_bounds__(256) void k_spsa(
    const float* __restrict__ EU, int invert, const float* __restrict__ wsp0,
    const float* __restrict__ bsp0, const _Float16* __restrict__ wsp1h,
    const float* __restrict__ bsp1, const _Float16* __restrict__ wsp2h,
    const float* __restrict__ bsp2, const float* __restrict__ wsa0,
    const float* __restrict__ bsa0, const float* __restrict__ wsa1,
    const float* __restrict__ bsa1, const float* __restrict__ wsa2,
    const float* __restrict__ bsa2, float* __restrict__ sp_out, float* __restrict__ sa_out) {
    int tile = blockIdx.x;
    int b = tile >> 10;
    int r = tile & 1023;
    int h = r >> 3;
    int w0 = (r & 7) << 4;

    __shared__ __attribute__((aligned(32))) _Float16 h0[16][128];
    __shared__ __attribute__((aligned(32))) _Float16 h1[16][128];
    __shared__ float Uloc[4][16];

    if (threadIdx.x < 64) {
        int q = threadIdx.x >> 4, n = threadIdx.x & 15;
        float u = EU[(size_t)(b * 4 + q) * HW + h * WW + w0 + n];
        Uloc[q][n] = invert ? (1.f - u) : u;
    }
    __syncthreads();

    // h0 = relu(W_sp0 @ U + b)
    for (int idx = threadIdx.x; idx < 2048; idx += 256) {
        int n = idx & 15, c = idx >> 4;
        float s = bsp0[c];
#pragma unroll
        for (int q = 0; q < 4; ++q) s += wsp0[c * 4 + q] * Uloc[q][n];
        h0[n][c] = (_Float16)fmaxf(s, 0.f);
    }
    __syncthreads();

    int wave = threadIdx.x >> 5;
    int lane = threadIdx.x & 31;
    int lm = lane & 15;
    int hi = lane >> 4;

    // GEMM1: h1 = relu(W_sp1 @ h0 + b)
    {
        v8f acc = {};
        const _Float16* Arow = wsp1h + (size_t)(wave * 16 + lm) * 128;
        for (int k0 = 0; k0 < 128; k0 += 32) {
            v16h a = load_a_frag(Arow, k0, hi);
            v16h bf = *(const v16h*)&h0[lm][k0 + hi * 16];
            acc = wmma_step(a, bf, acc);
        }
#pragma unroll
        for (int r2 = 0; r2 < 8; ++r2) {
            int ch = wave * 16 + hi * 8 + r2;
            h1[lm][ch] = (_Float16)fmaxf(acc[r2] + bsp1[ch], 0.f);
        }
    }
    __syncthreads();

    // GEMM2: sp = sigmoid(W_sp2 @ h1 + b)
    {
        v8f acc = {};
        const _Float16* Arow = wsp2h + (size_t)(wave * 16 + lm) * 128;
        for (int k0 = 0; k0 < 128; k0 += 32) {
            v16h a = load_a_frag(Arow, k0, hi);
            v16h bf = *(const v16h*)&h1[lm][k0 + hi * 16];
            acc = wmma_step(a, bf, acc);
        }
#pragma unroll
        for (int r2 = 0; r2 < 8; ++r2) {
            int ch = wave * 16 + hi * 8 + r2;
            float v = acc[r2] + bsp2[ch];
            sp_out[(size_t)(b * 128 + ch) * HW + h * WW + w0 + lm] = 1.f / (1.f + expf(-v));
        }
    }

    // sa chain (tiny): one pixel per thread for threads 0..15
    if (threadIdx.x < 16) {
        int n = threadIdx.x;
        int wx0 = w0 + n;
        float s1[9];
#pragma unroll
        for (int j = 0; j < 9; ++j) s1[j] = bsa0[j];
        for (int q = 0; q < 4; ++q) {
#pragma unroll
            for (int dy = -1; dy <= 1; ++dy) {
                int hh = h + dy;
                if (hh < 0 || hh >= HH) continue;
#pragma unroll
                for (int dx = -1; dx <= 1; ++dx) {
                    int wx = wx0 + dx;
                    if (wx < 0 || wx >= WW) continue;
                    float u = EU[(size_t)(b * 4 + q) * HW + hh * WW + wx];
                    if (invert) u = 1.f - u;
                    float* wbase = nullptr; (void)wbase;
#pragma unroll
                    for (int j = 0; j < 9; ++j)
                        s1[j] += wsa0[(j * 4 + q) * 9 + (dy + 1) * 3 + (dx + 1)] * u;
                }
            }
        }
        float s2[9];
#pragma unroll
        for (int j = 0; j < 9; ++j) {
            float t = bsa1[j];
#pragma unroll
            for (int q = 0; q < 9; ++q) t += wsa1[j * 9 + q] * fmaxf(s1[q], 0.f);
            s2[j] = fmaxf(t, 0.f);
        }
#pragma unroll
        for (int j = 0; j < 9; ++j) {
            float t = bsa2[j];
#pragma unroll
            for (int q = 0; q < 9; ++q) t += wsa2[j * 9 + q] * s2[q];
            sa_out[(size_t)(b * 9 + j) * HW + h * WW + wx0] = 1.f / (1.f + expf(-t));
        }
    }
}

// ---------------- fused uaconv + 1x1 conv + leaky-relu (WMMA) ----------------
// v[k=c*9+j] = patch(c,j) * sp[c] * sa[j];  y = Wdyn[64x1152] @ v;  out = leaky(Wc @ y + bc)
__global__ __launch_bounds__(128) void k_uaconv(
    const float* __restrict__ F1, const float* __restrict__ second,
    const float* __restrict__ sp, const float* __restrict__ sa,
    const _Float16* __restrict__ wdynh, const float* __restrict__ wc,
    const float* __restrict__ bc, float* __restrict__ outbuf) {
    int tile = blockIdx.x;
    int b = tile >> 10;
    int r = tile & 1023;
    int h = r >> 3;
    int w0 = (r & 7) << 4;

    __shared__ __attribute__((aligned(32))) _Float16 vb[16][1152];
    __shared__ float ytile[64][16];
    __shared__ float saL[9][16];
    __shared__ float spL[128][16];

    size_t pbase = (size_t)h * WW + w0;
    for (int idx = threadIdx.x; idx < 144; idx += 128) {
        int j = idx >> 4, n = idx & 15;
        saL[j][n] = sa[(size_t)(b * 9 + j) * HW + pbase + n];
    }
    for (int idx = threadIdx.x; idx < 2048; idx += 128) {
        int c = idx >> 4, n = idx & 15;
        spL[c][n] = sp[(size_t)(b * 128 + c) * HW + pbase + n];
    }
    __syncthreads();

    for (int idx = threadIdx.x; idx < 2048; idx += 128) {
        int c = idx >> 4, n = idx & 15;
        const float* src = (c < 64) ? (F1 + (size_t)(b * 64 + c) * HW)
                                    : (second + (size_t)(b * 64 + (c - 64)) * HW);
        float spv = spL[c][n];
        int wx0 = w0 + n;
#pragma unroll
        for (int j = 0; j < 9; ++j) {
            int dy = j / 3 - 1, dx = j % 3 - 1;
            int hh = h + dy, wx = wx0 + dx;
            float pv = (hh >= 0 && hh < HH && wx >= 0 && wx < WW) ? src[hh * WW + wx] : 0.f;
            vb[n][c * 9 + j] = (_Float16)(pv * spv * saL[j][n]);
        }
    }
    __syncthreads();

    int wave = threadIdx.x >> 5;   // 4 waves -> M tiles 0..3 (64 out channels)
    int lane = threadIdx.x & 31;
    int lm = lane & 15;
    int hi = lane >> 4;

    v8f acc = {};
    const _Float16* Arow = wdynh + (size_t)(wave * 16 + lm) * 1152;
    for (int k0 = 0; k0 < 1152; k0 += 32) {
        v16h a = load_a_frag(Arow, k0, hi);
        v16h bf = *(const v16h*)&vb[lm][k0 + hi * 16];
        acc = wmma_step(a, bf, acc);
    }
#pragma unroll
    for (int r2 = 0; r2 < 8; ++r2) ytile[wave * 16 + hi * 8 + r2][lm] = acc[r2];
    __syncthreads();

    for (int idx = threadIdx.x; idx < 1024; idx += 128) {
        int o = idx >> 4, n = idx & 15;
        float s = bc[o];
#pragma unroll 8
        for (int c = 0; c < 64; ++c) s += wc[o * 64 + c] * ytile[c][n];
        s = (s >= 0.f) ? s : 0.2f * s;
        outbuf[(size_t)(b * 64 + o) * HW + pbase + n] = s;
    }
}

// ---------------- launch ----------------
extern "C" void kernel_launch(void* const* d_in, const int* in_sizes, int n_in,
                              void* d_out, int out_size, void* d_ws, size_t ws_size,
                              hipStream_t stream) {
    const float* F_1 = (const float*)d_in[0];
    const float* ms = (const float*)d_in[1];
    const float* pan = (const float*)d_in[2];
    const int* masks = (const int*)d_in[3];
    const float* w_uec = (const float*)d_in[4];
    const float* b_uec = (const float*)d_in[5];
    const float* w_out = (const float*)d_in[6];
    const float* b_out = (const float*)d_in[7];
    const float* w_aue = (const float*)d_in[8];
    const float* b_aue = (const float*)d_in[9];
    const float* w_c1 = (const float*)d_in[10];
    const float* b_c1 = (const float*)d_in[11];
    const float* w_c2 = (const float*)d_in[12];
    const float* b_c2 = (const float*)d_in[13];
    // uac1: 14..26, uac2: 27..39  (w_sa0,b_sa0,w_sa1,b_sa1,w_sa2,b_sa2,
    //                              w_sp0,b_sp0,w_sp1,b_sp1,w_sp2,b_sp2,w_dyn)
    const int U1 = 14, U2 = 27;

    float* out = (float*)d_out;
    float* AU_out = out;                 // [2,4,H,W]
    float* EU_out = out + 131072;        // [2,4,H,W]
    float* mean_out = out + 262144;      // [2,4,H,W]
    float* F_out = out + 393216;         // [2,64,H,W]

    // workspace carve-up
    float* ws = (float*)d_ws;
    float* ws_x = ws;                    // 4194304  x = conv(F_1)
    float* ws_EU = ws_x + 4194304;       // 131072
    float* ws_sp = ws_EU + 131072;       // 2097152 reserved; sp is [2,128,HW] = 4194304
    float* ws_sa = ws_sp + 2097152;      // 294912
    float* ws_F2 = ws_sa + 294912;       // 2097152
    float* ws_wm = ws_F2 + 2097152;      // 5120
    _Float16* wec_h = (_Float16*)(ws_wm + 5120);   // 73728
    _Float16* wdyn1_h = wec_h + 73728;             // 73728
    _Float16* wdyn2_h = wdyn1_h + 73728;           // 73728
    _Float16* wsp1a_h = wdyn2_h + 73728;           // 16384
    _Float16* wsp2a_h = wsp1a_h + 16384;
    _Float16* wsp1b_h = wsp2a_h + 16384;
    _Float16* wsp2b_h = wsp1b_h + 16384;

    // --- prep: convert weights to f16, build masked W_out ---
    auto cvt = [&](const void* src, _Float16* dst, int n) {
        k_f32_to_f16<<<(n + 255) / 256, 256, 0, stream>>>((const float*)src, dst, n);
    };
    cvt(d_in[4], wec_h, 73728);
    cvt(d_in[U1 + 12], wdyn1_h, 73728);
    cvt(d_in[U2 + 12], wdyn2_h, 73728);
    cvt(d_in[U1 + 8], wsp1a_h, 16384);
    cvt(d_in[U1 + 10], wsp2a_h, 16384);
    cvt(d_in[U2 + 8], wsp1b_h, 16384);
    cvt(d_in[U2 + 10], wsp2b_h, 16384);
    k_wm<<<(5120 + 255) / 256, 256, 0, stream>>>(w_out, masks, ws_wm);

    // --- x = conv3x3(F_1) (WMMA implicit GEMM) ---
    k_conv_uec<<<NTILES, 256, 0, stream>>>(F_1, wec_h, b_uec, ws_x);

    // --- ensemble stats -> EU, mean ---
    k_stats<<<(NB * HW + 255) / 256, 256, 0, stream>>>(ws_x, ms, ws_wm, b_out, EU_out,
                                                       mean_out, ws_EU);

    // --- AU = sigmoid(conv3x3(x)) ---
    k_au<<<(NB * 4 * HW + 255) / 256, 256, 0, stream>>>(ws_x, w_aue, b_aue, AU_out);

    // --- uaconv pass 1: U = EU, x = [F_1, pan], then F_2 = leaky(conv1x1) ---
    k_spsa<<<NTILES, 256, 0, stream>>>(
        ws_EU, 0, (const float*)d_in[U1 + 6], (const float*)d_in[U1 + 7], wsp1a_h,
        (const float*)d_in[U1 + 9], wsp2a_h, (const float*)d_in[U1 + 11],
        (const float*)d_in[U1 + 0], (const float*)d_in[U1 + 1], (const float*)d_in[U1 + 2],
        (const float*)d_in[U1 + 3], (const float*)d_in[U1 + 4], (const float*)d_in[U1 + 5],
        ws_sp, ws_sa);
    k_uaconv<<<NTILES, 128, 0, stream>>>(F_1, pan, ws_sp, ws_sa, wdyn1_h, w_c1, b_c1, ws_F2);

    // --- uaconv pass 2: U = 1-EU, x = [F_1, F_2], then F_out = leaky(conv1x1) ---
    k_spsa<<<NTILES, 256, 0, stream>>>(
        ws_EU, 1, (const float*)d_in[U2 + 6], (const float*)d_in[U2 + 7], wsp1b_h,
        (const float*)d_in[U2 + 9], wsp2b_h, (const float*)d_in[U2 + 11],
        (const float*)d_in[U2 + 0], (const float*)d_in[U2 + 1], (const float*)d_in[U2 + 2],
        (const float*)d_in[U2 + 3], (const float*)d_in[U2 + 4], (const float*)d_in[U2 + 5],
        ws_sp, ws_sa);
    k_uaconv<<<NTILES, 128, 0, stream>>>(F_1, ws_F2, ws_sp, ws_sa, wdyn2_h, w_c2, b_c2,
                                         F_out);

    (void)in_sizes; (void)n_in; (void)out_size; (void)ws_size;
}
